// Surgery_ResidualAttentionBlock_26182120636696
// MI455X (gfx1250) — matrix operand order repaired and checked by reading, not compiled, not run on hardware
//
#include <hip/hip_runtime.h>

// ---------------------------------------------------------------------------
// Surgery ResidualAttentionBlock for MI455X (gfx1250, wave32, WMMA).
// Round 5: softmax cross-lane reductions via pure-VALU DPP16 butterflies
// (quad_perm/row_half_mirror/row_mirror); DPP control codes passed as
// template parameters (builtin requires frontend integer constants).
// ---------------------------------------------------------------------------

#define L_TOK  1024
#define BATCH  8
#define DMODEL 1024
#define NHEAD  16
#define HDIM   64

typedef __attribute__((ext_vector_type(16))) _Float16 v16h;
typedef __attribute__((ext_vector_type(8)))  _Float16 v8h;
typedef __attribute__((ext_vector_type(8)))  float    v8f;

// ---- DPP16 reductions over each 16-lane row (VALU only, no LDS) -----------
// Sequence: quad_perm(1,0,3,2)=0xB1, quad_perm(2,3,0,1)=0x4E,
// row_half_mirror=0x141, row_mirror=0x140.  Each stays within a 16-lane row.
template <int CTRL>
__device__ __forceinline__ float dpp_bcast(float x) {
  return __int_as_float(__builtin_amdgcn_update_dpp(
      __float_as_int(x), __float_as_int(x), CTRL, 0xf, 0xf, false));
}
__device__ __forceinline__ float row16_max(float x) {
  x = fmaxf(x, dpp_bcast<0xB1>(x));
  x = fmaxf(x, dpp_bcast<0x4E>(x));
  x = fmaxf(x, dpp_bcast<0x141>(x));
  x = fmaxf(x, dpp_bcast<0x140>(x));
  return x;
}
__device__ __forceinline__ float row16_sum(float x) {
  x += dpp_bcast<0xB1>(x);
  x += dpp_bcast<0x4E>(x);
  x += dpp_bcast<0x141>(x);
  x += dpp_bcast<0x140>(x);
  return x;
}

// ---- WMMA fragment loaders (layouts per CDNA5 ISA §7.12.2) -----

// A fragment (16x32 f16, M x K) from row-major f16 source.
// Lane = 16*khalf + m; element e -> K = e + 8*(e>=8) + 8*khalf.
__device__ __forceinline__ v16h load_a_frag(const _Float16* __restrict__ src, int lda) {
  const int lane = threadIdx.x & 31;
  const int m    = lane & 15;
  const int hi   = lane >> 4;
  const _Float16* p = src + (size_t)m * lda + (hi << 3);
  const v8h lo = *(const v8h*)(p);        // K = 8*hi + 0..7
  const v8h up = *(const v8h*)(p + 16);   // K = 16 + 8*hi + 0..7
  return __builtin_shufflevector(lo, up, 0, 1, 2, 3, 4, 5, 6, 7,
                                 8, 9, 10, 11, 12, 13, 14, 15);
}

// A fragment (16x32 f16) from an LDS tile stored row-major with ld=32.
__device__ __forceinline__ v16h load_a_frag_lds(const _Float16* __restrict__ src) {
  const int lane = threadIdx.x & 31;
  const int m    = lane & 15;
  const int hi   = lane >> 4;
  const _Float16* p = src + m * 32 + (hi << 3);
  const v8h lo = *(const v8h*)(p);
  const v8h up = *(const v8h*)(p + 16);
  return __builtin_shufflevector(lo, up, 0, 1, 2, 3, 4, 5, 6, 7,
                                 8, 9, 10, 11, 12, 13, 14, 15);
}

// B fragment (32x16, K x N) with B[k][n] = w[n*ldw + k]  (A * W^T, W row-major
// (N,K)).  Lane = 16*khalf + n; element e -> K = e + 8*(e>=8) + 8*khalf.
__device__ __forceinline__ v16h load_b_frag_nk(const _Float16* __restrict__ w, int ldw) {
  const int lane = threadIdx.x & 31;
  const int n    = lane & 15;
  const int hi   = lane >> 4;
  const _Float16* p = w + (size_t)n * ldw + (hi << 3);
  const v8h lo = *(const v8h*)(p);
  const v8h up = *(const v8h*)(p + 16);
  return __builtin_shufflevector(lo, up, 0, 1, 2, 3, 4, 5, 6, 7,
                                 8, 9, 10, 11, 12, 13, 14, 15);
}

// ---------------------------------------------------------------------------
// f32 -> f16 conversion (weights; 4 elements/thread, sizes are multiples of 4)
// ---------------------------------------------------------------------------
__global__ __launch_bounds__(256) void cvt_f16(const float* __restrict__ s,
                                               _Float16* __restrict__ d, int n) {
  const int i = (blockIdx.x * 256 + threadIdx.x) * 4;
  if (i < n) {
    const float4 f = *(const float4*)(s + i);
    d[i + 0] = (_Float16)f.x;
    d[i + 1] = (_Float16)f.y;
    d[i + 2] = (_Float16)f.z;
    d[i + 3] = (_Float16)f.w;
  }
}

// ---------------------------------------------------------------------------
// Transpose the V slice of qkv into head-major vT[b][h][d][n] so attention's
// P*V B-fragments become contiguous loads. Writes coalesced over n.
// ---------------------------------------------------------------------------
__global__ __launch_bounds__(256) void transpose_v(const _Float16* __restrict__ qkv,
                                                   _Float16* __restrict__ vT) {
  const int idx = blockIdx.x * 256 + threadIdx.x;   // over B*D*L, divisible
  const int n = idx & (L_TOK - 1);
  const int rest = idx >> 10;
  const int c = rest & (DMODEL - 1);                // c = h*64 + d
  const int b = rest >> 10;
  vT[idx] = qkv[((size_t)n * BATCH + b) * (3 * DMODEL) + 2 * DMODEL + c];
}

// ---------------------------------------------------------------------------
// Custom LayerNorm over axis 1 (the BATCH axis of (L,B,D)); gamma/beta are
// indexed by the L axis (reference broadcasts w[:,None,None] with L==D).
// ---------------------------------------------------------------------------
__global__ __launch_bounds__(256) void ln_axis1(const float* __restrict__ x,
                                                const float* __restrict__ w,
                                                const float* __restrict__ bp,
                                                _Float16* __restrict__ y) {
  const int idx = blockIdx.x * 256 + threadIdx.x;  // over L*D (exactly divisible)
  const int l = idx >> 10;          // / DMODEL
  const int d = idx & 1023;         // % DMODEL
  const float* xp = x + (size_t)l * BATCH * DMODEL + d;
  float mean = 0.f;
#pragma unroll
  for (int bb = 0; bb < BATCH; ++bb) mean += xp[bb * DMODEL];
  mean *= (1.0f / BATCH);
  float var = 0.f;
#pragma unroll
  for (int bb = 0; bb < BATCH; ++bb) {
    const float t = xp[bb * DMODEL] - mean;
    var += t * t;
  }
  var *= (1.0f / BATCH);
  const float inv = rsqrtf(var + 1e-6f);
  const float g = w[l], be = bp[l];
  _Float16* yp = y + (size_t)l * BATCH * DMODEL + d;
#pragma unroll
  for (int bb = 0; bb < BATCH; ++bb)
    yp[bb * DMODEL] = (_Float16)(g * ((xp[bb * DMODEL] - mean) * inv) + be);
}

// ---------------------------------------------------------------------------
// WMMA GEMM:  C[M,N] = act(A[M,K] * W[N,K]^T + bias) (+ residual)
// 8 waves/block; each wave owns a 16x128 tile (8 accumulators, A reused 8x).
// ACT: 0 identity, 1 QuickGELU.  OUT16: write f16 (activations) vs f32.
// ---------------------------------------------------------------------------
template <int ACT, int OUT16>
__global__ __launch_bounds__(256) void gemm_wmma(const _Float16* __restrict__ A,
                                                 const _Float16* __restrict__ W,
                                                 const float* __restrict__ bias,
                                                 const float* __restrict__ residual,
                                                 void* __restrict__ Cout,
                                                 int M, int N, int K) {
  const int wid  = threadIdx.x >> 5;
  const int lane = threadIdx.x & 31;
  const int nt128 = N >> 7;
  const int tile = blockIdx.x * 8 + wid;            // tile counts divide by 8
  const int mt = tile / nt128;
  const int nt = tile % nt128;

  const _Float16* Arow = A + (size_t)mt * 16 * K;
  const _Float16* Wbase = W + (size_t)nt * 128 * K;
  v8f acc[8] = {};

  for (int k0 = 0; k0 < K; k0 += 32) {
    __builtin_prefetch(Arow + k0 + 128, 0, 1);       // global_prefetch_b8
    const v16h a = load_a_frag(Arow + k0, K);
#pragma unroll
    for (int t = 0; t < 8; ++t) {
      const v16h b = load_b_frag_nk(Wbase + (size_t)t * 16 * K + k0, K);
      acc[t] = __builtin_amdgcn_wmma_f32_16x16x32_f16(
          false, a, false, b, (short)0, acc[t], false, false);
    }
  }

  // Epilogue. C layout: lane(0-15)->N=lane,M=j ; lane(16-31)->N=lane-16,M=j+8
  const int hi  = lane >> 4;
  const int nlo = lane & 15;
#pragma unroll
  for (int t = 0; t < 8; ++t) {
    const int col = nt * 128 + t * 16 + nlo;
    const float bval = bias ? bias[col] : 0.0f;
#pragma unroll
    for (int j = 0; j < 8; ++j) {
      const int row = mt * 16 + j + hi * 8;
      float v = acc[t][j] + bval;
      if (ACT == 1) v = v / (1.0f + __expf(-1.702f * v));  // QuickGELU
      if (residual) v += residual[(size_t)row * N + col];
      if (OUT16) ((_Float16*)Cout)[(size_t)row * N + col] = (_Float16)v;
      else       ((float*)Cout)[(size_t)row * N + col] = v;
    }
  }
}

// ---------------------------------------------------------------------------
// Flash-attention style kernel (f16 in / f16 out, f32 softmax state).
// One wave per (b, h, 16-query tile); streams 32-key blocks.
// qkv layout: qkv[(n*B + b)*3D + sel + h*64 + d], sel in {0, D, 2D}.
// vT layout:  vT[((b*H + h)*64 + d)*L + n]   (d-major per head).
// ---------------------------------------------------------------------------
__global__ __launch_bounds__(256) void attn_wmma(const _Float16* __restrict__ qkv,
                                                 const _Float16* __restrict__ vT,
                                                 int qoff, int koff,
                                                 _Float16* __restrict__ out) {
  __shared__ _Float16 pbuf[8][16 * 32];              // per-wave P staging (8KB)
  const int wid  = threadIdx.x >> 5;
  const int lane = threadIdx.x & 31;
  const int id = blockIdx.x * 8 + wid;               // B*H*(L/16) ids, /8 exact
  const int qt = id & 63;                            // query tile (L/16 = 64)
  const int bh = id >> 6;
  const int h  = bh & (NHEAD - 1);
  const int b  = bh >> 4;

  const int    ld   = BATCH * 3 * DMODEL;            // stride between seq rows
  const size_t base = (size_t)b * 3 * DMODEL + (size_t)h * HDIM;
  const _Float16* Q  = qkv + qoff + base;
  const _Float16* Kp = qkv + koff + base;
  const _Float16* Vt = vT + (size_t)(b * NHEAD + h) * HDIM * L_TOK;

  // Q tile (16 x 64) as two A fragments (K 0-31 / 32-63), loaded once.
  const v16h qa0 = load_a_frag(Q + (size_t)qt * 16 * ld, ld);
  const v16h qa1 = load_a_frag(Q + (size_t)qt * 16 * ld + 32, ld);

  v8f o[4] = {};
  float rmax[8], rsum[8];
#pragma unroll
  for (int j = 0; j < 8; ++j) { rmax[j] = -1e30f; rsum[j] = 0.0f; }
  const int hi  = lane >> 4;
  const int nlo = lane & 15;
  const float scale = 0.125f;                        // hd^-0.5, hd = 64

  for (int kb = 0; kb < L_TOK; kb += 32) {
    const _Float16* Kblk = Kp + (size_t)kb * ld;
    // S (16q x 32k): two 16-key halves, K-dim 64 accumulated over 2 WMMA.
    v8f s[2];
#pragma unroll
    for (int nh = 0; nh < 2; ++nh) {
      v8f c = {};
      const v16h bk0 = load_b_frag_nk(Kblk + (size_t)nh * 16 * ld, ld);
      c = __builtin_amdgcn_wmma_f32_16x16x32_f16(false, qa0, false, bk0, (short)0, c, false, false);
      const v16h bk1 = load_b_frag_nk(Kblk + (size_t)nh * 16 * ld + 32, ld);
      c = __builtin_amdgcn_wmma_f32_16x16x32_f16(false, qa1, false, bk1, (short)0, c, false, false);
      s[nh] = c;
    }

    // Online softmax. Row m = j + 8*hi lives in the 16 lanes of this half;
    // reductions are pure-VALU DPP16 butterflies (no LDS traffic).
    float p0v[8], p1v[8];
#pragma unroll
    for (int j = 0; j < 8; ++j) {
      const float s0 = s[0][j] * scale;
      const float s1 = s[1][j] * scale;
      const float m = row16_max(fmaxf(s0, s1));
      const float mnew  = fmaxf(rmax[j], m);
      const float alpha = __expf(rmax[j] - mnew);
      const float p0 = __expf(s0 - mnew);
      const float p1 = __expf(s1 - mnew);
      const float psum = row16_sum(p0 + p1);
      rsum[j] = rsum[j] * alpha + psum;
      rmax[j] = mnew;
#pragma unroll
      for (int t = 0; t < 4; ++t) o[t][j] *= alpha;
      p0v[j] = p0; p1v[j] = p1;
    }

    // Transpose P (C-fragment layout -> A-fragment layout) through LDS.
    // Same-wave DS ops are in-order (ISA §7.3), no explicit wait required.
    _Float16* pl = pbuf[wid];
#pragma unroll
    for (int j = 0; j < 8; ++j) {
      const int row = j + hi * 8;
      pl[row * 32 + nlo]      = (_Float16)p0v[j];
      pl[row * 32 + 16 + nlo] = (_Float16)p1v[j];
    }
    const v16h pa = load_a_frag_lds(pl);

    // O (16 x 64) += P (16 x 32) * V (32 x 64).
    // vT is d-major, so B[k][n] = Vt[(t*16+n)*L + kb + k]: contiguous loads.
#pragma unroll
    for (int t = 0; t < 4; ++t) {
      const v16h vb = load_b_frag_nk(Vt + (size_t)(t * 16) * L_TOK + kb, L_TOK);
      o[t] = __builtin_amdgcn_wmma_f32_16x16x32_f16(
          false, pa, false, vb, (short)0, o[t], false, false);
    }
  }

  // Write O / rowsum (f16) to (L,B,D) layout: out[(n*B+b)*D + h*64 + d].
  _Float16* obase = out + (size_t)b * DMODEL + (size_t)h * HDIM;
#pragma unroll
  for (int t = 0; t < 4; ++t) {
#pragma unroll
    for (int j = 0; j < 8; ++j) {
      const int row = qt * 16 + j + hi * 8;
      const int col = t * 16 + nlo;
      obase[(size_t)row * BATCH * DMODEL + col] = (_Float16)(o[t][j] / rsum[j]);
    }
  }
}

// ---------------------------------------------------------------------------
extern "C" void kernel_launch(void* const* d_in, const int* in_sizes, int n_in,
                              void* d_out, int out_size, void* d_ws, size_t ws_size,
                              hipStream_t stream) {
  const float* x       = (const float*)d_in[0];
  const float* ln1_w   = (const float*)d_in[1];
  const float* ln1_b   = (const float*)d_in[2];
  const float* ln2_w   = (const float*)d_in[3];
  const float* ln2_b   = (const float*)d_in[4];
  const float* qkv_w   = (const float*)d_in[5];
  const float* proj_w  = (const float*)d_in[6];
  const float* proj_b  = (const float*)d_in[7];
  const float* fc_w    = (const float*)d_in[8];
  const float* fc_b    = (const float*)d_in[9];
  const float* cproj_w = (const float*)d_in[10];
  const float* cproj_b = (const float*)d_in[11];
  float* out = (float*)d_out;

  const size_t LBD = (size_t)L_TOK * BATCH * DMODEL;   // 8.39M elements
  const size_t DD  = (size_t)DMODEL * DMODEL;          // 1.05M

  // Workspace carving (f16 activations, f16 weights, f32 residual).
  char* p = (char*)d_ws;
  _Float16* xln   = (_Float16*)p; p += LBD * 2;        // ln1 out, reused for ln2
  _Float16* qkv   = (_Float16*)p; p += 3 * LBD * 2;
  _Float16* vT    = (_Float16*)p; p += LBD * 2;        // transposed V
  _Float16* a_ori = (_Float16*)p; p += LBD * 2;
  _Float16* a_new = (_Float16*)p; p += LBD * 2;
  _Float16* fcout = (_Float16*)p; p += 4 * LBD * 2;
  float*    x_ori = (float*)p;    p += LBD * 4;
  _Float16* qkvw_h  = (_Float16*)p; p += 3 * DD * 2;
  _Float16* projw_h = (_Float16*)p; p += DD * 2;
  _Float16* fcw_h   = (_Float16*)p; p += 4 * DD * 2;
  _Float16* cprojw_h= (_Float16*)p; p += 4 * DD * 2;

  const int M = L_TOK * BATCH;   // 8192 token rows
  const dim3 blk(256);

  // 0) Convert weights to f16 (L2-resident thereafter).
  cvt_f16<<<(int)(3 * DD / 1024), blk, 0, stream>>>(qkv_w,   qkvw_h,  (int)(3 * DD));
  cvt_f16<<<(int)(DD / 1024),     blk, 0, stream>>>(proj_w,  projw_h, (int)DD);
  cvt_f16<<<(int)(4 * DD / 1024), blk, 0, stream>>>(fc_w,    fcw_h,   (int)(4 * DD));
  cvt_f16<<<(int)(4 * DD / 1024), blk, 0, stream>>>(cproj_w, cprojw_h,(int)(4 * DD));

  // 1) LN1 over batch axis (f16 out)
  ln_axis1<<<(L_TOK * DMODEL) / 256, blk, 0, stream>>>(x, ln1_w, ln1_b, xln);

  // 2) QKV projection: (8192 x 1024) * (1024 x 3072), f16 out
  int tiles = (M / 16) * ((3 * DMODEL) / 128);
  gemm_wmma<0, 1><<<tiles / 8, blk, 0, stream>>>(xln, qkvw_h, nullptr, nullptr,
                                                 qkv, M, 3 * DMODEL, DMODEL);

  // 2b) head-major transpose of V for the P*V WMMA operand
  transpose_v<<<(int)(LBD / 256), blk, 0, stream>>>(qkv, vT);

  // 3/4) attention: original softmax(qk^T)v and surgery softmax(vv^T)v
  const int ablocks = BATCH * NHEAD * (L_TOK / 16) / 8;
  attn_wmma<<<ablocks, blk, 0, stream>>>(qkv, vT, 0, DMODEL, a_ori);
  attn_wmma<<<ablocks, blk, 0, stream>>>(qkv, vT, 2 * DMODEL, 2 * DMODEL, a_new);

  // 5) x_new = x + proj(a_new)  -> d_out[0 : LBD), f32
  tiles = (M / 16) * (DMODEL / 128);
  gemm_wmma<0, 0><<<tiles / 8, blk, 0, stream>>>(a_new, projw_h, proj_b, x,
                                                 out, M, DMODEL, DMODEL);
  // 6) x_ori = x + proj(a_ori), f32
  gemm_wmma<0, 0><<<tiles / 8, blk, 0, stream>>>(a_ori, projw_h, proj_b, x,
                                                 x_ori, M, DMODEL, DMODEL);

  // 7) LN2 over batch axis (f16 out, reuses xln)
  ln_axis1<<<(L_TOK * DMODEL) / 256, blk, 0, stream>>>(x_ori, ln2_w, ln2_b, xln);

  // 8) fc + QuickGELU: (8192 x 1024) * (1024 x 4096), f16 out
  tiles = (M / 16) * ((4 * DMODEL) / 128);
  gemm_wmma<1, 1><<<tiles / 8, blk, 0, stream>>>(xln, fcw_h, fc_b, nullptr,
                                                 fcout, M, 4 * DMODEL, DMODEL);

  // 9) d_out[LBD : 2LBD) = x_ori + cproj(h), f32
  tiles = (M / 16) * (DMODEL / 128);
  gemm_wmma<0, 0><<<tiles / 8, blk, 0, stream>>>(fcout, cprojw_h, cproj_b, x_ori,
                                                 out + LBD, M, DMODEL, 4 * DMODEL);
}
